// RowInteraction_85418309583186
// MI455X (gfx1250) — matrix operand order repaired and checked by reading, not compiled, not run on hardware
//
#include <hip/hip_runtime.h>
#include <math.h>

typedef __attribute__((ext_vector_type(16))) __bf16 v16bf;
typedef __attribute__((ext_vector_type(8)))  __bf16 v8bf;
typedef __attribute__((ext_vector_type(8)))  float  v8f;

#define NSPc     6
#define WINDOWc  4
#define NEGc     (-1000000000.0f)
#define Ec       128
#define Fc       512
#define NHEADc   8
#define DHc      16
#define MAX_LPAD 160

static __device__ __forceinline__ __bf16 f2bf(float f) {
  unsigned u = __builtin_bit_cast(unsigned, f);
  unsigned r = (u + 0x7FFFu + ((u >> 16) & 1u)) >> 16;
  unsigned short s = (unsigned short)r;
  return __builtin_bit_cast(__bf16, s);
}
static __device__ __forceinline__ float bf2f(__bf16 b) {
  unsigned short s = __builtin_bit_cast(unsigned short, b);
  unsigned u = ((unsigned)s) << 16;
  return __builtin_bit_cast(float, u);
}
static __device__ __forceinline__ float gelu_approx(float x) {
  const float k0 = 0.7978845608028654f; // sqrt(2/pi)
  float x3 = x * x * x;
  return 0.5f * x * (1.0f + tanhf(k0 * (x + 0.044715f * x3)));
}

// ---- transpose + convert weights: dst[n*K+k] = bf16(src[k*N+n]) ----
__global__ void k_transpose_bf16(const float* __restrict__ src,
                                 __bf16* __restrict__ dst, int K, int N) {
  int idx = blockIdx.x * 256 + threadIdx.x;
  if (idx >= K * N) return;
  int n = idx / K, k = idx - n * K;
  dst[idx] = f2bf(src[(size_t)k * N + n]);
}

// ---- build per-scale sequence: specials + grouped means ----
__global__ void k_build_x(const float* __restrict__ emb, const float* __restrict__ cls,
                          const float* __restrict__ glob, float* __restrict__ X,
                          int L, int g, int total) {
  int idx = blockIdx.x * 256 + threadIdx.x;
  if (idx >= total) return;
  int e = idx & 127;
  int t = idx >> 7;                // token index
  int bt = t / L, l = t - bt * L;
  float v;
  if (l < NSPc) {
    v = (l < 4) ? cls[l * Ec + e] : glob[(l - 4) * Ec + e];
  } else {
    int kf = l - NSPc;
    const float* p = emb + ((size_t)bt * (NSPc + 128) + NSPc + (size_t)kf * g) * Ec + e;
    float s = 0.f;
    for (int j = 0; j < g; j++) s += p[(size_t)j * Ec];
    v = s / (float)g;
  }
  X[idx] = v;
}

// ---- layernorm f32 -> bf16, one wave per token ----
__global__ void k_ln(const float* __restrict__ X, const float* __restrict__ gm,
                     const float* __restrict__ bt, __bf16* __restrict__ H, int N) {
  int w = threadIdx.x >> 5, lane = threadIdx.x & 31;
  int tok = blockIdx.x * 8 + w;
  if (tok >= N) return;
  const float* x = X + (size_t)tok * Ec + lane * 4;
  float4 v = *(const float4*)x;
  float s = v.x + v.y + v.z + v.w;
  for (int off = 16; off; off >>= 1) s += __shfl_xor(s, off, 32);
  float mu = s * (1.0f / 128.0f);
  float d0 = v.x - mu, d1 = v.y - mu, d2 = v.z - mu, d3 = v.w - mu;
  float q = d0 * d0 + d1 * d1 + d2 * d2 + d3 * d3;
  for (int off = 16; off; off >>= 1) q += __shfl_xor(q, off, 32);
  float rs = rsqrtf(q * (1.0f / 128.0f) + 1e-5f);
  int e = lane * 4;
  __bf16* h = H + (size_t)tok * Ec + e;
  h[0] = f2bf(d0 * rs * gm[e + 0] + bt[e + 0]);
  h[1] = f2bf(d1 * rs * gm[e + 1] + bt[e + 1]);
  h[2] = f2bf(d2 * rs * gm[e + 2] + bt[e + 2]);
  h[3] = f2bf(d3 * rs * gm[e + 3] + bt[e + 3]);
}

// ---- bf16 WMMA GEMM: out = act(A[N,K] @ Wt[Nout,K]^T + bias) (+residual f32) ----
// mode 0: store bf16; mode 1: GELU + store bf16; mode 2: f32 residual add
__global__ void k_gemm(const __bf16* __restrict__ A, const __bf16* __restrict__ Wt,
                       const float* __restrict__ bias, __bf16* __restrict__ outb,
                       float* __restrict__ outf, int K, int Nout, int mode) {
  int lane = threadIdx.x & 31;
  int w    = threadIdx.x >> 5;
  int row0 = blockIdx.x * 16;
  int c0   = blockIdx.y * 128 + w * 16;
  int arow = row0 + (lane & 15);
  int kbA  = (lane >> 4) * 8;
  int bn   = c0 + (lane & 15);
  int kbB  = (lane >> 4) * 8;
  const __bf16* ab = A  + (size_t)arow * K + kbA;
  const __bf16* bb = Wt + (size_t)bn   * K + kbB;
  v8f acc;
#pragma unroll
  for (int i = 0; i < 8; i++) acc[i] = 0.f;
  for (int k0 = 0; k0 < K; k0 += 32) {
    v8bf alo = *(const v8bf*)(ab + k0);
    v8bf ahi = *(const v8bf*)(ab + k0 + 16);
    v8bf blo = *(const v8bf*)(bb + k0);
    v8bf bhi = *(const v8bf*)(bb + k0 + 16);
    v16bf a, b;
#pragma unroll
    for (int i = 0; i < 8; i++) {
      a[i] = alo[i]; a[i + 8] = ahi[i];
      b[i] = blo[i]; b[i + 8] = bhi[i];
    }
    acc = __builtin_amdgcn_wmma_f32_16x16x32_bf16(false, a, false, b,
                                                  (short)0, acc, false, false);
  }
  float bv = bias[bn];
  int mbase = row0 + ((lane >> 4) << 3);
  if (mode == 2) {
#pragma unroll
    for (int r = 0; r < 8; r++) {
      size_t o = (size_t)(mbase + r) * Nout + bn;
      outf[o] += acc[r] + bv;
    }
  } else if (mode == 1) {
#pragma unroll
    for (int r = 0; r < 8; r++) {
      size_t o = (size_t)(mbase + r) * Nout + bn;
      outb[o] = f2bf(gelu_approx(acc[r] + bv));
    }
  } else {
#pragma unroll
    for (int r = 0; r < 8; r++) {
      size_t o = (size_t)(mbase + r) * Nout + bn;
      outb[o] = f2bf(acc[r] + bv);
    }
  }
}

// ---- RoPE in-place on bf16 Q or K ----
__global__ void k_rope(__bf16* __restrict__ Q, int N, int L) {
  int idx = blockIdx.x * 256 + threadIdx.x;
  if (idx >= N * 64) return;
  int d = idx & 7;
  int h = (idx >> 3) & 7;
  int n = idx >> 6;
  int pos = n % L;
  float inv = powf(100000.0f, -(float)d * 0.125f);
  float ang = (float)pos * inv;
  float sn = sinf(ang), cs = cosf(ang);
  size_t base = (size_t)n * Ec + h * DHc + d;
  float x1 = bf2f(Q[base]);
  float x2 = bf2f(Q[base + 8]);
  Q[base]     = f2bf(x1 * cs - x2 * sn);
  Q[base + 8] = f2bf(x1 * sn + x2 * cs);
}

// ---- attention: one wave per (seq, head, 16-row tile); WMMA both matmuls ----
__global__ void k_attn(const __bf16* __restrict__ Q, const __bf16* __restrict__ Kb,
                       const __bf16* __restrict__ V, __bf16* __restrict__ O,
                       int L, int Lpad) {
  __shared__ float  sc[16 * MAX_LPAD];
  __shared__ __bf16 sa[16 * MAX_LPAD];
  int lane = threadIdx.x;
  int m0   = blockIdx.x * 16;
  int h    = blockIdx.y;
  int seq  = blockIdx.z;
  size_t sbase = (size_t)seq * L * Ec + h * DHc;
  const __bf16 z = f2bf(0.0f);
  int nl  = lane & 15;
  int hiL = lane >> 4;
  int kbA = hiL * 8;

  // A fragment: 16 q rows, K = DH padded 16 -> 32 with zeros
  int mr = m0 + nl; if (mr >= L) mr = L - 1;
  v16bf aq;
  {
    const __bf16* p = Q + sbase + (size_t)mr * Ec + kbA;
    v8bf qv = *(const v8bf*)p;
#pragma unroll
    for (int i = 0; i < 8; i++) { aq[i] = qv[i]; aq[i + 8] = z; }
  }

  // scores vs all column tiles
  for (int ct = 0; ct * 16 < Lpad; ct++) {
    v16bf bk;
    {
      int nr = ct * 16 + nl; if (nr >= L) nr = L - 1;
      const __bf16* p = Kb + sbase + (size_t)nr * Ec + kbA;
      v8bf k0v = *(const v8bf*)p;
#pragma unroll
      for (int i = 0; i < 8; i++) { bk[i] = k0v[i]; bk[i + 8] = z; }
    }
    v8f dacc;
#pragma unroll
    for (int i = 0; i < 8; i++) dacc[i] = 0.f;
    dacc = __builtin_amdgcn_wmma_f32_16x16x32_bf16(false, aq, false, bk,
                                                   (short)0, dacc, false, false);
    int mloc = hiL * 8;
    int j_g  = ct * 16 + nl;
#pragma unroll
    for (int r = 0; r < 8; r++) {
      int i_g = m0 + mloc + r;
      float v = dacc[r] * 0.25f; // 1/sqrt(DH)
      if (j_g >= L) {
        v = NEGc;
      } else {
        int diff = i_g - j_g; if (diff < 0) diff = -diff;
        bool allowed = (diff <= WINDOWc) || (i_g < NSPc) || (j_g < NSPc);
        if (!allowed) v += NEGc;
      }
      sc[(mloc + r) * Lpad + j_g] = v;
    }
  }
  __syncthreads();

  // row softmax in LDS, write bf16 probabilities
  for (int rm = 0; rm < 16; rm++) {
    float mx = -3.4e38f;
    for (int c = lane; c < Lpad; c += 32) mx = fmaxf(mx, sc[rm * Lpad + c]);
    for (int off = 16; off; off >>= 1) mx = fmaxf(mx, __shfl_xor(mx, off, 32));
    float sum = 0.f;
    for (int c = lane; c < Lpad; c += 32) {
      float e = expf(sc[rm * Lpad + c] - mx);
      sc[rm * Lpad + c] = e;
      sum += e;
    }
    for (int off = 16; off; off >>= 1) sum += __shfl_xor(sum, off, 32);
    float inv = 1.0f / sum;
    for (int c = lane; c < Lpad; c += 32)
      sa[rm * Lpad + c] = f2bf(sc[rm * Lpad + c] * inv);
  }
  __syncthreads();

  // attn @ V
  v8f oacc;
#pragma unroll
  for (int i = 0; i < 8; i++) oacc[i] = 0.f;
  int kbB = hiL * 8;
  for (int kt = 0; kt < Lpad; kt += 32) {
    v16bf a, b;
#pragma unroll
    for (int i = 0; i < 8; i++) {
      a[i]     = sa[nl * Lpad + kt + kbA + i];
      a[i + 8] = sa[nl * Lpad + kt + kbA + 16 + i];
    }
#pragma unroll
    for (int i = 0; i < 16; i++) {
      int r = kt + kbB + i + (i >= 8 ? 8 : 0); if (r >= L) r = L - 1;
      b[i] = V[sbase + (size_t)r * Ec + nl];
    }
    oacc = __builtin_amdgcn_wmma_f32_16x16x32_bf16(false, a, false, b,
                                                   (short)0, oacc, false, false);
  }
  int mloc = hiL * 8;
#pragma unroll
  for (int r = 0; r < 8; r++) {
    int m = m0 + mloc + r;
    if (m < L) O[sbase + (size_t)m * Ec + nl] = f2bf(oacc[r]);
  }
}

// ---- final: mean of CLS rows over scales + output layernorm ----
__global__ void k_final(const float* __restrict__ X0, const float* __restrict__ X1,
                        const float* __restrict__ X2, const float* __restrict__ og,
                        const float* __restrict__ ob, float* __restrict__ out,
                        int L0, int L1, int L2) {
  int w = threadIdx.x >> 5, lane = threadIdx.x & 31;
  int row = blockIdx.x * 8 + w;  // 0..1023
  int bt = row >> 2, c = row & 3;
  int e = lane * 4;
  float4 a  = *(const float4*)(X0 + ((size_t)bt * L0 + c) * Ec + e);
  float4 b1 = *(const float4*)(X1 + ((size_t)bt * L1 + c) * Ec + e);
  float4 c1 = *(const float4*)(X2 + ((size_t)bt * L2 + c) * Ec + e);
  float v0 = (a.x + b1.x + c1.x) * (1.0f / 3.0f);
  float v1 = (a.y + b1.y + c1.y) * (1.0f / 3.0f);
  float v2 = (a.z + b1.z + c1.z) * (1.0f / 3.0f);
  float v3 = (a.w + b1.w + c1.w) * (1.0f / 3.0f);
  float s = v0 + v1 + v2 + v3;
  for (int off = 16; off; off >>= 1) s += __shfl_xor(s, off, 32);
  float mu = s * (1.0f / 128.0f);
  float d0 = v0 - mu, d1 = v1 - mu, d2 = v2 - mu, d3 = v3 - mu;
  float q = d0 * d0 + d1 * d1 + d2 * d2 + d3 * d3;
  for (int off = 16; off; off >>= 1) q += __shfl_xor(q, off, 32);
  float rs = rsqrtf(q * (1.0f / 128.0f) + 1e-5f);
  float* o = out + (size_t)bt * 512 + c * 128 + e;
  o[0] = d0 * rs * og[e + 0] + ob[e + 0];
  o[1] = d1 * rs * og[e + 1] + ob[e + 1];
  o[2] = d2 * rs * og[e + 2] + ob[e + 2];
  o[3] = d3 * rs * og[e + 3] + ob[e + 3];
}

extern "C" void kernel_launch(void* const* d_in, const int* in_sizes, int n_in,
                              void* d_out, int out_size, void* d_ws, size_t ws_size,
                              hipStream_t stream) {
  (void)in_sizes; (void)n_in; (void)out_size; (void)ws_size;
  const float* emb  = (const float*)d_in[0];
  const float* cls  = (const float*)d_in[1];
  const float* glob = (const float*)d_in[2];
  const float* Wq   = (const float*)d_in[3];
  const float* bq   = (const float*)d_in[4];
  const float* Wk   = (const float*)d_in[5];
  const float* bk   = (const float*)d_in[6];
  const float* Wv   = (const float*)d_in[7];
  const float* bv   = (const float*)d_in[8];
  const float* Wo   = (const float*)d_in[9];
  const float* bo   = (const float*)d_in[10];
  const float* ln1g = (const float*)d_in[11];
  const float* ln1b = (const float*)d_in[12];
  const float* ln2g = (const float*)d_in[13];
  const float* ln2b = (const float*)d_in[14];
  const float* W1   = (const float*)d_in[15];
  const float* b1   = (const float*)d_in[16];
  const float* W2   = (const float*)d_in[17];
  const float* b2   = (const float*)d_in[18];
  const float* outg = (const float*)d_in[19];
  const float* outbp= (const float*)d_in[20];

  const int BT = 256;
  const int Ls[3]    = {134, 38, 22};
  const int Gs[3]    = {1, 4, 8};
  const int Lpads[3] = {160, 64, 32};
  const int NMAX = BT * 134;

  char* ws = (char*)d_ws;
  size_t off = 0;
  auto alloc = [&](size_t bytes) -> char* {
    char* p = ws + off;
    off += (bytes + 255) & ~(size_t)255;
    return p;
  };
  float* X[3];
  for (int s = 0; s < 3; s++) X[s] = (float*)alloc((size_t)BT * Ls[s] * Ec * 4);
  __bf16* H    = (__bf16*)alloc((size_t)NMAX * Ec * 2);
  __bf16* qkvo = (__bf16*)alloc((size_t)4 * NMAX * Ec * 2);  // Q,K,V,O; aliased by G
  __bf16* Qb   = qkvo;
  __bf16* Kbuf = qkvo + (size_t)NMAX * Ec;
  __bf16* Vbuf = qkvo + (size_t)2 * NMAX * Ec;
  __bf16* Ob   = qkvo + (size_t)3 * NMAX * Ec;
  __bf16* G    = qkvo;  // FFN intermediate [NMAX,512] aliases dead Q/K/V/O
  __bf16* WT   = (__bf16*)alloc((size_t)6 * 196608 * 2);

  // ---- weight prep: transpose + bf16 convert ----
  for (int sl = 0; sl < 6; sl++) {
    __bf16* base = WT + (size_t)sl * 196608;
    k_transpose_bf16<<<64, 256, 0, stream>>>(Wq + (size_t)sl * 16384, base,           128, 128);
    k_transpose_bf16<<<64, 256, 0, stream>>>(Wk + (size_t)sl * 16384, base + 16384,   128, 128);
    k_transpose_bf16<<<64, 256, 0, stream>>>(Wv + (size_t)sl * 16384, base + 32768,   128, 128);
    k_transpose_bf16<<<64, 256, 0, stream>>>(Wo + (size_t)sl * 16384, base + 49152,   128, 128);
    k_transpose_bf16<<<256, 256, 0, stream>>>(W1 + (size_t)sl * 65536, base + 65536,  128, 512);
    k_transpose_bf16<<<256, 256, 0, stream>>>(W2 + (size_t)sl * 65536, base + 131072, 512, 128);
  }

  // ---- build per-scale sequences ----
  for (int s = 0; s < 3; s++) {
    int N = BT * Ls[s], tot = N * Ec;
    k_build_x<<<(tot + 255) / 256, 256, 0, stream>>>(emb, cls, glob, X[s], Ls[s], Gs[s], tot);
  }

  // ---- encoder layers ----
  for (int s = 0; s < 3; s++) {
    int L = Ls[s], Lpad = Lpads[s], N = BT * L;
    int MT = (L + 15) / 16;
    for (int l = 0; l < 2; l++) {
      int sl = s * 2 + l;
      const __bf16* wt = WT + (size_t)sl * 196608;
      dim3 g16(N / 16, 1), g512(N / 16, 4);

      k_ln<<<(N + 7) / 8, 256, 0, stream>>>(X[s], ln1g + sl * 128, ln1b + sl * 128, H, N);
      k_gemm<<<g16, 256, 0, stream>>>(H, wt,          bq + sl * 128, Qb,   nullptr, 128, 128, 0);
      k_gemm<<<g16, 256, 0, stream>>>(H, wt + 16384,  bk + sl * 128, Kbuf, nullptr, 128, 128, 0);
      k_gemm<<<g16, 256, 0, stream>>>(H, wt + 32768,  bv + sl * 128, Vbuf, nullptr, 128, 128, 0);
      k_rope<<<(N * 64 + 255) / 256, 256, 0, stream>>>(Qb,   N, L);
      k_rope<<<(N * 64 + 255) / 256, 256, 0, stream>>>(Kbuf, N, L);
      k_attn<<<dim3(MT, NHEADc, BT), 32, 0, stream>>>(Qb, Kbuf, Vbuf, Ob, L, Lpad);
      k_gemm<<<g16, 256, 0, stream>>>(Ob, wt + 49152, bo + sl * 128, nullptr, X[s], 128, 128, 2);
      k_ln<<<(N + 7) / 8, 256, 0, stream>>>(X[s], ln2g + sl * 128, ln2b + sl * 128, H, N);
      k_gemm<<<g512, 256, 0, stream>>>(H, wt + 65536,  b1 + sl * 512, G, nullptr, 128, 512, 1);
      k_gemm<<<g16,  256, 0, stream>>>(G, wt + 131072, b2 + sl * 128, nullptr, X[s], 512, 128, 2);
    }
  }

  // ---- CLS mean over scales + output LN ----
  k_final<<<128, 256, 0, stream>>>(X[0], X[1], X[2], outg, outbp, (float*)d_out,
                                   Ls[0], Ls[1], Ls[2]);
}
